// FGHGNNConvLayer_40862318854642
// MI455X (gfx1250) — compile-verified, matrix operand order checked by reading, not run on hardware
//
#include <hip/hip_runtime.h>

// ---------------------------------------------------------------------------
// FGHGNN hetero conv layer for gfx1250 (MI455X).
// Dense linear algebra -> v_wmma_f32_16x16x32_f16 (fp32 acc), LDS-staged with
// B pre-transposed so both A and B fragments load as 2x ds_load_b128.
// f32->f16 staging packs via v_cvt_pk_rtz_f16_f32 (one VALU per dword).
// Edge scatter/softmax -> VALU + global float atomics (HBM-bound anyway).
// ---------------------------------------------------------------------------

typedef __attribute__((ext_vector_type(16))) _Float16 v16h;
typedef __attribute__((ext_vector_type(2)))  __fp16   v2fp16;  // cvt_pkrtz return type
typedef __attribute__((ext_vector_type(8)))  float    v8f;

#define DIM 128
#define NH  4
#define HD  512
#define BNI 0.9999950000374997f   // 1/sqrt(1+1e-5), eval-mode BatchNorm

// ---- helpers ---------------------------------------------------------------

__device__ __forceinline__ unsigned pack2h(float lo, float hi) {
    union { v2fp16 h; unsigned u; } t;
    t.h = __builtin_amdgcn_cvt_pkrtz(lo, hi);   // v_cvt_pk_rtz_f16_f32
    return t.u;
}

__device__ __forceinline__ float f4get(const float4& v, int j) {
    return j == 0 ? v.x : j == 1 ? v.y : j == 2 ? v.z : v.w;
}

union V16U { uint4 q[2]; v16h v; };

// ========================= WMMA tiled GEMM =================================
// Out[M,N] = epi( A[M,K] @ B[K,N] )
//   epi: x = acc (+bias[n]) (+Cadd[m,n]); HAS_BN: x = relu(bn_g[n]*x*BNI+bn_b[n])
//        ACCUM ? Out += x : Out = x
// Block: 256 threads (8 waves, wave32). Tile 64(M) x 128(N), BK=32.
// Wave grid 2x4, each wave 32x32 = 2x2 wmma accumulators.
// K-loop software-pipelined: next tile's global loads overlap current compute.
template<bool HAS_BIAS, bool HAS_CADD, bool HAS_BN, bool ACCUM>
__global__ __launch_bounds__(256)
void wmma_gemm_kernel(const float* __restrict__ A, const float* __restrict__ B,
                      const float* __restrict__ Cadd, float* __restrict__ Out,
                      int M, int N, int K,
                      const float* __restrict__ bias,
                      const float* __restrict__ bn_g, const float* __restrict__ bn_b)
{
    alignas(16) __shared__ _Float16 As[64][32];    // [m][k]   4 KB
    alignas(16) __shared__ _Float16 BsT[128][32];  // [n][k]   8 KB (transposed)

    const int tid  = threadIdx.x;
    const int wid  = tid >> 5;
    const int lane = tid & 31;
    const int wm   = wid >> 2;        // 0..1
    const int wn   = wid & 3;         // 0..3
    const int bm   = blockIdx.x * 64;
    const int bn   = blockIdx.y * 128;

    v8f acc[2][2] = {};

    // A staging: thread -> row (0..63), 8 consecutive k (2 float4 loads)
    const int arow = tid >> 2;
    const int acol = (tid & 3) * 8;
    const bool arow_ok = (bm + arow) < M;
    // B staging: thread -> k-pair (0..15 -> k=2kp,2kp+1), 8 consecutive n
    const int kp  = tid & 15;
    const int n8  = (tid >> 4) * 8;   // 0,8,...,120

    const int r0 = lane & 15;
    const int kb = (lane >> 4) * 8;   // 0 or 8 (ISA 16-bit frag K split)

    float4 a0, a1, br0a, br0b, br1a, br1b;

    auto load_tile = [&](int k0) {
        const float4 z4 = make_float4(0.f, 0.f, 0.f, 0.f);
        if (arow_ok) {
            const float4* ap = (const float4*)(A + (size_t)(bm + arow) * K + (k0 + acol));
            a0 = ap[0]; a1 = ap[1];
        } else { a0 = z4; a1 = z4; }
        const float4* b0p = (const float4*)(B + (size_t)(k0 + 2 * kp) * N + (bn + n8));
        const float4* b1p = (const float4*)(B + (size_t)(k0 + 2 * kp + 1) * N + (bn + n8));
        br0a = b0p[0]; br0b = b0p[1];
        br1a = b1p[0]; br1b = b1p[1];
    };

    auto store_tile = [&]() {
        uint4 w;
        w.x = pack2h(a0.x, a0.y); w.y = pack2h(a0.z, a0.w);
        w.z = pack2h(a1.x, a1.y); w.w = pack2h(a1.z, a1.w);
        *(uint4*)&As[arow][acol] = w;                      // one ds_store_b128
#pragma unroll
        for (int j = 0; j < 8; ++j) {                      // eight ds_store_b32
            float lo = (j < 4) ? f4get(br0a, j) : f4get(br0b, j - 4);
            float hi = (j < 4) ? f4get(br1a, j) : f4get(br1b, j - 4);
            *(unsigned*)&BsT[n8 + j][2 * kp] = pack2h(lo, hi);
        }
    };

    load_tile(0);
    for (int k0 = 0; k0 < K; k0 += 32) {
        store_tile();
        __syncthreads();
        if (k0 + 32 < K) load_tile(k0 + 32);               // overlap with compute

        v16h bf[2];
#pragma unroll
        for (int ni = 0; ni < 2; ++ni) {
            const _Float16* bp = &BsT[wn * 32 + ni * 16 + r0][0];
            V16U u;
            u.q[0] = *(const uint4*)(bp + kb);             // ds_load_b128
            u.q[1] = *(const uint4*)(bp + kb + 16);        // ds_load_b128
            bf[ni] = u.v;
        }
#pragma unroll
        for (int mi = 0; mi < 2; ++mi) {
            const _Float16* ap = &As[wm * 32 + mi * 16 + r0][0];
            V16U u;
            u.q[0] = *(const uint4*)(ap + kb);
            u.q[1] = *(const uint4*)(ap + kb + 16);
            v16h af = u.v;
#pragma unroll
            for (int ni = 0; ni < 2; ++ni) {
                acc[mi][ni] = __builtin_amdgcn_wmma_f32_16x16x32_f16(
                    false, af, false, bf[ni], (short)0, acc[mi][ni], false, false);
            }
        }
        __syncthreads();
    }

    // ---- epilogue: C/D layout row = v + 8*(lane>=16), col = lane&15
    const int rt = 8 * (lane >> 4);
    const int ct = lane & 15;

    auto epilogue = [&](bool guard) {
#pragma unroll
        for (int ni = 0; ni < 2; ++ni) {
            const int colg = bn + wn * 32 + ni * 16 + ct;
            float bi = 0.0f, g = 0.0f, be = 0.0f;
            if (HAS_BIAS) bi = bias[colg];
            if (HAS_BN) { g = bn_g[colg]; be = bn_b[colg]; }
#pragma unroll
            for (int mi = 0; mi < 2; ++mi) {
                const int rbase = bm + wm * 32 + mi * 16 + rt;
#pragma unroll
                for (int v = 0; v < 8; ++v) {
                    const int rowg = rbase + v;
                    if (!guard || rowg < M) {
                        const size_t idx = (size_t)rowg * N + colg;
                        float x = acc[mi][ni][v];
                        if (HAS_BIAS) x += bi;
                        if (HAS_CADD) x += Cadd[idx];
                        if (HAS_BN)   x = fmaxf(g * x * BNI + be, 0.0f);
                        if (ACCUM) Out[idx] += x; else Out[idx] = x;
                    }
                }
            }
        }
    };
    if (bm + 64 <= M) epilogue(false);   // fast path, no per-row guards
    else              epilogue(true);    // partial last M-block
}

// ========================= elementwise / scatter kernels ====================

__global__ void fill_kernel(float* __restrict__ p, long long n, float v) {
    long long i = (long long)blockIdx.x * blockDim.x + threadIdx.x;
    if (i < n) p[i] = v;
}

// neigh[dst] += relu(x[src] + ef[e]); one wave per edge, 4 floats per lane
__global__ void gine_scatter_kernel(const float* __restrict__ x, const float* __restrict__ ef,
                                    const int* __restrict__ src, const int* __restrict__ dst,
                                    float* __restrict__ neigh, int E)
{
    int e = (int)(((long long)blockIdx.x * blockDim.x + threadIdx.x) >> 5);
    int lane = threadIdx.x & 31;
    if (e >= E) return;
    int s = src[e], d = dst[e];
    float4 a = ((const float4*)(x  + (size_t)s * DIM))[lane];
    float4 b = ((const float4*)(ef + (size_t)e * DIM))[lane];
    float* nd = neigh + (size_t)d * DIM + lane * 4;
    atomicAdd(nd + 0, fmaxf(a.x + b.x, 0.0f));
    atomicAdd(nd + 1, fmaxf(a.y + b.y, 0.0f));
    atomicAdd(nd + 2, fmaxf(a.z + b.z, 0.0f));
    atomicAdd(nd + 3, fmaxf(a.w + b.w, 0.0f));
}

// h <- (1+eps)*x + h    (in-place on the neigh buffer)
__global__ void gine_h0_kernel(const float* __restrict__ x, const float* __restrict__ eps,
                               float* __restrict__ h, long long n)
{
    long long i = (long long)blockIdx.x * blockDim.x + threadIdx.x;
    if (i < n) h[i] = (1.0f + eps[0]) * x[i] + h[i];
}

// out[k,h] = sum_d Wfc[k, h*D+d] * avec[h, d]   (512 outputs)
__global__ void attvec_kernel(const float* __restrict__ Wfc, const float* __restrict__ avec,
                              float* __restrict__ out)
{
    int t = blockIdx.x * blockDim.x + threadIdx.x;
    if (t >= DIM * NH) return;
    int k = t >> 2, h = t & 3;
    float s = 0.0f;
    const float* wr = Wfc + (size_t)k * HD + h * DIM;
    const float* av = avec + h * DIM;
    for (int d = 0; d < DIM; ++d) s += wr[d] * av[d];
    out[k * NH + h] = s;
}

// out[n,h] = sum_k x[n,k] * vvec[k,h]
__global__ void attscore_kernel(const float* __restrict__ x, const float* __restrict__ vvec,
                                float* __restrict__ out, int N)
{
    int t = blockIdx.x * blockDim.x + threadIdx.x;
    if (t >= N * NH) return;
    int n = t >> 2, h = t & 3;
    const float* xr = x + (size_t)n * DIM;
    float s = 0.0f;
    for (int k = 0; k < DIM; ++k) s += xr[k] * vvec[k * NH + h];
    out[t] = s;
}

__device__ __forceinline__ void atomicMaxF(float* addr, float val) {
    unsigned int* ua = (unsigned int*)addr;
    unsigned int old = *ua;
    while (__uint_as_float(old) < val) {
        unsigned int assumed = old;
        old = atomicCAS(ua, assumed, __float_as_uint(val));
        if (old == assumed) break;
    }
}

// e = leaky_relu(el[src]+er[dst]); esc[e,h]=e; emax[dst,h]=max(...)
__global__ void edge_logits_kernel(const float* __restrict__ el, const float* __restrict__ er,
                                   const int* __restrict__ src, const int* __restrict__ dst,
                                   float* __restrict__ esc, float* __restrict__ emax, int E)
{
    int e = blockIdx.x * blockDim.x + threadIdx.x;
    if (e >= E) return;
    int s = src[e], d = dst[e];
#pragma unroll
    for (int h = 0; h < NH; ++h) {
        float v = el[s * NH + h] + er[d * NH + h];
        v = v > 0.0f ? v : 0.2f * v;
        esc[(size_t)e * NH + h] = v;
        atomicMaxF(&emax[d * NH + h], v);
    }
}

// esc <- exp(esc - emax[dst]); den[dst] += esc
__global__ void edge_exp_kernel(float* __restrict__ esc, const float* __restrict__ emax,
                                float* __restrict__ den, const int* __restrict__ dst, int E)
{
    int e = blockIdx.x * blockDim.x + threadIdx.x;
    if (e >= E) return;
    int d = dst[e];
#pragma unroll
    for (int h = 0; h < NH; ++h) {
        float ex = __expf(esc[(size_t)e * NH + h] - emax[d * NH + h]);
        esc[(size_t)e * NH + h] = ex;
        atomicAdd(&den[d * NH + h], ex);
    }
}

// rst[dst, h*D+d] += (esc/den) * hs[src, h*D+d]; one wave per edge, 16 f/lane
__global__ void edge_apply_kernel(const float* __restrict__ esc, const float* __restrict__ den,
                                  const float* __restrict__ hs,
                                  const int* __restrict__ src, const int* __restrict__ dst,
                                  float* __restrict__ rst, int E)
{
    int e = (int)(((long long)blockIdx.x * blockDim.x + threadIdx.x) >> 5);
    int lane = threadIdx.x & 31;
    if (e >= E) return;
    int s = src[e], d = dst[e];
    int h = lane >> 3;                       // 16 contiguous floats => head = lane/8
    float a = esc[(size_t)e * NH + h] / fmaxf(den[(size_t)d * NH + h], 1e-9f);
    const float4* hv = (const float4*)(hs + (size_t)s * HD) + lane * 4;
    float* rd = rst + (size_t)d * HD + lane * 16;
#pragma unroll
    for (int j = 0; j < 4; ++j) {
        float4 v = hv[j];
        atomicAdd(rd + j * 4 + 0, a * v.x);
        atomicAdd(rd + j * 4 + 1, a * v.y);
        atomicAdd(rd + j * 4 + 2, a * v.z);
        atomicAdd(rd + j * 4 + 3, a * v.w);
    }
}

// ========================= host orchestration ===============================

static inline void launch_gemm(hipStream_t st, const float* A, const float* B,
                               const float* Cadd, float* Out, int M, int N, int K,
                               const float* bias, const float* bg, const float* bb,
                               int accum)
{
    dim3 grid((M + 63) / 64, N / 128);
    if (bg && Cadd)
        wmma_gemm_kernel<true, true, true, false><<<grid, 256, 0, st>>>(A, B, Cadd, Out, M, N, K, bias, bg, bb);
    else if (bg)
        wmma_gemm_kernel<true, false, true, false><<<grid, 256, 0, st>>>(A, B, Cadd, Out, M, N, K, bias, bg, bb);
    else if (accum)
        wmma_gemm_kernel<true, false, false, true><<<grid, 256, 0, st>>>(A, B, Cadd, Out, M, N, K, bias, bg, bb);
    else if (bias)
        wmma_gemm_kernel<true, false, false, false><<<grid, 256, 0, st>>>(A, B, Cadd, Out, M, N, K, bias, bg, bb);
    else
        wmma_gemm_kernel<false, false, false, false><<<grid, 256, 0, st>>>(A, B, Cadd, Out, M, N, K, bias, bg, bb);
}

extern "C" void kernel_launch(void* const* d_in, const int* in_sizes, int n_in,
                              void* d_out, int out_size, void* d_ws, size_t ws_size,
                              hipStream_t stream)
{
    // ---- inputs (setup_inputs order)
    const float* x_atom   = (const float*)d_in[0];
    const float* x_group  = (const float*)d_in[1];
    const float* ef_bond  = (const float*)d_in[2];
    const float* ef_over  = (const float*)d_in[3];
    const int* bond_src = (const int*)d_in[4];  const int* bond_dst = (const int*)d_in[5];
    const int* ov_src   = (const int*)d_in[6];  const int* ov_dst   = (const int*)d_in[7];
    const int* ref_src  = (const int*)d_in[8];  const int* ref_dst  = (const int*)d_in[9];
    const int* pool_src = (const int*)d_in[10]; const int* pool_dst = (const int*)d_in[11];

    const float* b_eps = (const float*)d_in[12];
    const float* b_W1  = (const float*)d_in[13]; const float* b_b1  = (const float*)d_in[14];
    const float* b_g1  = (const float*)d_in[15]; const float* b_be1 = (const float*)d_in[16];
    const float* b_W2  = (const float*)d_in[17]; const float* b_b2  = (const float*)d_in[18];
    const float* o_eps = (const float*)d_in[19];
    const float* o_W1  = (const float*)d_in[20]; const float* o_b1  = (const float*)d_in[21];
    const float* o_g1  = (const float*)d_in[22]; const float* o_be1 = (const float*)d_in[23];
    const float* o_W2  = (const float*)d_in[24]; const float* o_b2  = (const float*)d_in[25];
    const float* r_Wfc = (const float*)d_in[26]; const float* r_al  = (const float*)d_in[27];
    const float* r_ar  = (const float*)d_in[28]; const float* r_Wres= (const float*)d_in[29];
    const float* r_bias= (const float*)d_in[30]; const float* r_g   = (const float*)d_in[31];
    const float* r_b   = (const float*)d_in[32]; const float* r_Wout= (const float*)d_in[33];
    const float* r_bout= (const float*)d_in[34];
    const float* p_Wfc = (const float*)d_in[35]; const float* p_al  = (const float*)d_in[36];
    const float* p_ar  = (const float*)d_in[37]; const float* p_Wres= (const float*)d_in[38];
    const float* p_bias= (const float*)d_in[39]; const float* p_g   = (const float*)d_in[40];
    const float* p_b   = (const float*)d_in[41]; const float* p_Wout= (const float*)d_in[42];
    const float* p_bout= (const float*)d_in[43];

    const int NA  = in_sizes[0] / DIM;     // 100000
    const int NG  = in_sizes[1] / DIM;     // 25000
    const int E_B = in_sizes[4];           // 400000
    const int E_O = in_sizes[6];           // 100000
    const int E_R = in_sizes[8];           // 200000
    const int E_P = in_sizes[10];          // 200000

    float* out_atom  = (float*)d_out;
    float* out_group = out_atom + (size_t)NA * DIM;

    // ---- workspace carve (reused across sequential phases)
    float* ws = (float*)d_ws; size_t off = 0;
    auto carve = [&](size_t n) { float* p = ws + off; off += n; return p; };
    float* bufA = carve((size_t)NA * HD);   // h1_atom / h1_group / hs_p
    float* bufB = carve((size_t)NA * HD);   // rst_r -> flat_r (in place)
    float* bufC = carve((size_t)NA * DIM);  // neigh/h0 (atom, then group)
    float* bufD = carve((size_t)NG * HD);   // hs_r, then rst_p -> flat_p
    float* sb0  = carve((size_t)NA * NH);   // er_r / el_p
    float* sb1  = carve((size_t)NA * NH);   // emax_r
    float* sb2  = carve((size_t)NA * NH);   // den_r
    float* sb3  = carve((size_t)NG * NH);   // el_r / er_p
    float* sb4  = carve((size_t)NG * NH);   // emax_p
    float* sb5  = carve((size_t)NG * NH);   // den_p
    int EMX = E_R > E_P ? E_R : E_P;
    float* esc  = carve((size_t)EMX * NH);  // edge scores (reused)
    float* valv = carve(DIM * NH);          // src-side attention vector
    float* varv = carve(DIM * NH);          // dst-side attention vector

    const float NEG_INF = -3.0e38f;
    auto blocks = [](long long n, int bs) { return (unsigned)((n + bs - 1) / bs); };

    // ================= Phase 1: GINE(bond) -> atoms =================
    (void)hipMemsetAsync(bufC, 0, (size_t)NA * DIM * sizeof(float), stream);
    gine_scatter_kernel<<<blocks((long long)E_B * 32, 256), 256, 0, stream>>>(
        x_atom, ef_bond, bond_src, bond_dst, bufC, E_B);
    gine_h0_kernel<<<blocks((long long)NA * DIM, 256), 256, 0, stream>>>(
        x_atom, b_eps, bufC, (long long)NA * DIM);
    launch_gemm(stream, bufC, b_W1, nullptr, bufA, NA, 2 * DIM, DIM, b_b1, b_g1, b_be1, 0);
    launch_gemm(stream, bufA, b_W2, nullptr, out_atom, NA, DIM, 2 * DIM, b_b2, nullptr, nullptr, 0);

    // ================= Phase 2: GAT(refine) group -> atoms ==========
    attvec_kernel<<<2, 256, 0, stream>>>(r_Wfc, r_al, valv);
    attvec_kernel<<<2, 256, 0, stream>>>(r_Wfc, r_ar, varv);
    launch_gemm(stream, x_group, r_Wfc, nullptr, bufD, NG, HD, DIM, nullptr, nullptr, nullptr, 0); // hs_r
    attscore_kernel<<<blocks((long long)NG * NH, 256), 256, 0, stream>>>(x_group, valv, sb3, NG);
    attscore_kernel<<<blocks((long long)NA * NH, 256), 256, 0, stream>>>(x_atom,  varv, sb0, NA);
    fill_kernel<<<blocks((long long)NA * NH, 256), 256, 0, stream>>>(sb1, (long long)NA * NH, NEG_INF);
    edge_logits_kernel<<<blocks(E_R, 256), 256, 0, stream>>>(sb3, sb0, ref_src, ref_dst, esc, sb1, E_R);
    (void)hipMemsetAsync(sb2, 0, (size_t)NA * NH * sizeof(float), stream);
    edge_exp_kernel<<<blocks(E_R, 256), 256, 0, stream>>>(esc, sb1, sb2, ref_dst, E_R);
    (void)hipMemsetAsync(bufB, 0, (size_t)NA * HD * sizeof(float), stream);
    edge_apply_kernel<<<blocks((long long)E_R * 32, 256), 256, 0, stream>>>(
        esc, sb2, bufD, ref_src, ref_dst, bufB, E_R);
    launch_gemm(stream, x_atom, r_Wres, bufB, bufB, NA, HD, DIM, r_bias, r_g, r_b, 0);  // +BN+ReLU
    launch_gemm(stream, bufB, r_Wout, nullptr, out_atom, NA, DIM, HD, r_bout, nullptr, nullptr, 1);

    // ================= Phase 3: GINE(overlap) -> groups =============
    (void)hipMemsetAsync(bufC, 0, (size_t)NG * DIM * sizeof(float), stream);
    gine_scatter_kernel<<<blocks((long long)E_O * 32, 256), 256, 0, stream>>>(
        x_group, ef_over, ov_src, ov_dst, bufC, E_O);
    gine_h0_kernel<<<blocks((long long)NG * DIM, 256), 256, 0, stream>>>(
        x_group, o_eps, bufC, (long long)NG * DIM);
    launch_gemm(stream, bufC, o_W1, nullptr, bufA, NG, 2 * DIM, DIM, o_b1, o_g1, o_be1, 0);
    launch_gemm(stream, bufA, o_W2, nullptr, out_group, NG, DIM, 2 * DIM, o_b2, nullptr, nullptr, 0);

    // ================= Phase 4: GAT(pool) atoms -> groups ===========
    attvec_kernel<<<2, 256, 0, stream>>>(p_Wfc, p_al, valv);
    attvec_kernel<<<2, 256, 0, stream>>>(p_Wfc, p_ar, varv);
    launch_gemm(stream, x_atom, p_Wfc, nullptr, bufA, NA, HD, DIM, nullptr, nullptr, nullptr, 0);  // hs_p
    attscore_kernel<<<blocks((long long)NA * NH, 256), 256, 0, stream>>>(x_atom,  valv, sb0, NA);
    attscore_kernel<<<blocks((long long)NG * NH, 256), 256, 0, stream>>>(x_group, varv, sb3, NG);
    fill_kernel<<<blocks((long long)NG * NH, 256), 256, 0, stream>>>(sb4, (long long)NG * NH, NEG_INF);
    edge_logits_kernel<<<blocks(E_P, 256), 256, 0, stream>>>(sb0, sb3, pool_src, pool_dst, esc, sb4, E_P);
    (void)hipMemsetAsync(sb5, 0, (size_t)NG * NH * sizeof(float), stream);
    edge_exp_kernel<<<blocks(E_P, 256), 256, 0, stream>>>(esc, sb4, sb5, pool_dst, E_P);
    (void)hipMemsetAsync(bufD, 0, (size_t)NG * HD * sizeof(float), stream);
    edge_apply_kernel<<<blocks((long long)E_P * 32, 256), 256, 0, stream>>>(
        esc, sb5, bufA, pool_src, pool_dst, bufD, E_P);
    launch_gemm(stream, x_group, p_Wres, bufD, bufD, NG, HD, DIM, p_bias, p_g, p_b, 0);
    launch_gemm(stream, bufD, p_Wout, nullptr, out_group, NG, DIM, HD, p_bout, nullptr, nullptr, 1);
}
